// LSTM_13348758356539
// MI455X (gfx1250) — matrix operand order repaired and checked by reading, not compiled, not run on hardware
//
#include <hip/hip_runtime.h>
#include <cmath>

#define B_SZ 256
#define T_SZ 512

typedef __attribute__((ext_vector_type(16))) _Float16 v16h;
typedef __attribute__((ext_vector_type(8)))  float    v8f;
typedef __attribute__((ext_vector_type(4)))  unsigned int v4u;
typedef __attribute__((ext_vector_type(8)))  int      v8i;
typedef __attribute__((ext_vector_type(4)))  int      v4i;

#if __has_builtin(__builtin_amdgcn_tensor_load_to_lds) && \
    __has_builtin(__builtin_amdgcn_s_wait_tensorcnt)
#define USE_TDM 1
#else
#define USE_TDM 0
#endif

__device__ __forceinline__ v8f wmma16x16x32(v16h a, v16h b, v8f c) {
  // D = A(16x32 f16) * B(32x16 f16) + C(16x16 f32)
  return __builtin_amdgcn_wmma_f32_16x16x32_f16(false, a, false, b, (short)0, c,
                                                false, false);
}

__device__ __forceinline__ float sigmoidf_(float x) {
  return 1.0f / (1.0f + __expf(-x));
}

// ---- fragment index maps (ISA 7.12.2, wave32) ----
// A 16x32 f16: lane holds 16 halves. lanes 0-15: K=0..7,16..23 ; lanes 16-31: K=8..15,24..31
__device__ __forceinline__ int afrag_idx(int m, int kk) {
  int hi = (kk >> 3) & 1;                 // odd 8-group -> upper lane half
  int lane = m + (hi << 4);
  int e = (kk < 8) ? kk : (kk < 16 ? kk - 8 : (kk < 24 ? kk - 8 : kk - 16));
  return lane * 16 + e;
}
// B 32x16 f16: lanes 0-15 hold K=0..7,16..23 of column N=lane; lanes 16-31 hold K=8..15,24..31
__device__ __forceinline__ int bfrag_idx(int nl, int kk) {
  return (nl + (((kk >> 3) & 1) << 4)) * 16 + ((kk & 7) + ((kk >> 4) << 3));
}

// ============================================================================
// Input projection: xg[r, n] = sum_k x[r, k] * W[n, k] + bias[n]
// rows = B*T. Each block: 128 rows (8 waves x 16-row M-tiles), all G columns.
// Weights staged once per block into LDS as pre-swizzled B fragments.
// ============================================================================
template <int DIN, int G>
__global__ void gemm_in_kernel(const float* __restrict__ x,
                               const float* __restrict__ W,
                               const float* __restrict__ bias,
                               float* __restrict__ xg) {
  constexpr int KPAD = (DIN + 31) & ~31;
  constexpr int KC = KPAD / 32;
  constexpr int NT = G / 16;
  extern __shared__ char smem[];
  _Float16* ldsB = (_Float16*)smem;                          // KPAD*G halves
  float*    ldsX = (float*)(smem + (size_t)KPAD * G * 2);    // 128*KPAD floats
  const int tid = threadIdx.x;
  const int wave = tid >> 5, lane = tid & 31;
  const int row0 = blockIdx.x * 128;

  // stage weights as B fragments (zero-padded K)
  for (int idx = tid; idx < G * KPAD; idx += 256) {
    int n = idx / KPAD, k = idx - n * KPAD;
    float w = (k < DIN) ? W[n * DIN + k] : 0.0f;
    ldsB[((n >> 4) * KC + (k >> 5)) * 512 + bfrag_idx(n & 15, k & 31)] =
        (_Float16)w;
  }
  // stage 128 activation rows (zero-padded K), coalesced
  for (int idx = tid; idx < 128 * KPAD; idx += 256) {
    int r = idx / KPAD, k = idx - r * KPAD;
    ldsX[idx] = (k < DIN) ? x[(size_t)(row0 + r) * DIN + k] : 0.0f;
  }
  __syncthreads();

  // build this wave's A fragments (rows mrow..mrow+15)
  const int mrow = wave * 16;
  const int am = lane & 15;
  const int kbase = (lane >> 4) * 8;
  v16h afr[KC];
#pragma unroll
  for (int kc = 0; kc < KC; ++kc) {
#pragma unroll
    for (int e = 0; e < 16; ++e) {
      int k = (e < 8) ? (kbase + e) : (16 + kbase + (e - 8));
      afr[kc][e] = (_Float16)ldsX[(mrow + am) * KPAD + kc * 32 + k];
    }
  }

  const int nl = lane & 15;
  const int mb = (lane >> 4) * 8;
  for (int nt = 0; nt < NT; ++nt) {
    int n = nt * 16 + nl;
    float bv = bias[n];
    v8f acc;
#pragma unroll
    for (int j = 0; j < 8; ++j) acc[j] = bv;
#pragma unroll
    for (int kc = 0; kc < KC; ++kc) {
      v16h bfr = *(const v16h*)&ldsB[(nt * KC + kc) * 512 + lane * 16];
      acc = wmma16x16x32(afr[kc], bfr, acc);
    }
#pragma unroll
    for (int j = 0; j < 8; ++j)
      xg[(size_t)(row0 + mrow + mb + j) * G + n] = acc[j];
  }
}

// ============================================================================
// Persistent recurrent scan: grid (B/16, 2 dirs). Each block owns 16 batch
// rows for all T steps. Whh resident in LDS as B fragments; h kept in LDS in
// A-fragment layout (written by elementwise stage); c in registers.
// Gate tile (16 x 4H input pre-activations) is double-buffered in LDS and
// fetched one step ahead by the Tensor Data Mover (TENSORcnt-tracked), so the
// per-step global latency is fully hidden behind WMMA + elementwise work.
// ============================================================================
template <int H>
__global__ void scan_kernel(const float* __restrict__ xgF,
                            const float* __restrict__ xgB,
                            const float* __restrict__ WhhF,
                            const float* __restrict__ WhhB,
                            float* __restrict__ out) {
  constexpr int G = 4 * H;
  constexpr int KC = H / 32;
  constexpr int NT = G / 16;
  constexpr int TPW = NT / 8;           // N-tiles per wave (8 waves)
  constexpr int CPT = (16 * H) / 256;   // cell-states per thread
  extern __shared__ char smem[];
  _Float16* ldsB = (_Float16*)smem;                                 // H*G halves
  _Float16* ldsA = (_Float16*)(smem + (size_t)H * G * 2);           // KC*512
  float*    ldsG0 = (float*)(smem + (size_t)H * G * 2 + KC * 1024); // 2 x 16*G
  const int tid = threadIdx.x;
  const int wave = tid >> 5, lane = tid & 31;
  const int dir = blockIdx.y;
  const int b0 = blockIdx.x * 16;
  const float* xg  = dir ? xgB : xgF;
  const float* Whh = dir ? WhhB : WhhF;

  // stage Whh^T as B fragments: B[k][n] = Whh[n*H + k]
  for (int idx = tid; idx < G * H; idx += 256) {
    int n = idx / H, k = idx - n * H;
    ldsB[((n >> 4) * KC + (k >> 5)) * 512 + bfrag_idx(n & 15, k & 31)] =
        (_Float16)Whh[idx];
  }
  for (int idx = tid; idx < KC * 512; idx += 256) ldsA[idx] = (_Float16)0.0f;
  float c_state[CPT];
#pragma unroll
  for (int q = 0; q < CPT; ++q) c_state[q] = 0.0f;
  const int pm = tid >> 4;     // elementwise row
  const int pj0 = tid & 15;

#if USE_TDM
  // Issue one 2D TDM descriptor: tile = 16 rows x G floats, row stride T*G.
  auto issue_tile = [&](int tt, float* buf) {
    const unsigned long long ga =
        (unsigned long long)(uintptr_t)(xg + ((size_t)b0 * T_SZ + tt) * G);
    const unsigned int la = (unsigned int)(uintptr_t)buf;
    v4u g0;
    g0[0] = 1u;                                     // count=1 (valid D#)
    g0[1] = la;                                     // lds_addr
    g0[2] = (unsigned int)ga;                       // global_addr[31:0]
    g0[3] = (unsigned int)(ga >> 32) | (2u << 30);  // global_addr[56:32] | type=2
    const unsigned int td0 = (unsigned int)G;       // tensor_dim0 (elems)
    const unsigned int td1 = 16u;                   // tensor_dim1 (rows)
    const unsigned long long st0 = (unsigned long long)T_SZ * G;  // row stride
    v8i g1;
    g1[0] = (int)(2u << 16);                        // data_size=4B, no flags
    g1[1] = (int)((td0 & 0xffffu) << 16);           // tensor_dim0[15:0]
    g1[2] = (int)((td0 >> 16) | ((td1 & 0xffffu) << 16));
    g1[3] = (int)((td1 >> 16) | ((unsigned int)G << 16));  // tile_dim0=G
    g1[4] = 16;                                     // tile_dim1=16, tile_dim2=0
    g1[5] = (int)(unsigned int)(st0 & 0xffffffffu); // dim0_stride[31:0]
    g1[6] = (int)(unsigned int)((st0 >> 32) & 0xffffu);
    g1[7] = 0;
    v4i g2 = {0, 0, 0, 0};
    v4i g3 = {0, 0, 0, 0};
    v8i g4 = {0, 0, 0, 0, 0, 0, 0, 0};
    // amdgpu-toolchain (clang-23) 6-arg form: (g0, g1, g2, g3, g4, cpol)
    __builtin_amdgcn_tensor_load_to_lds(g0, g1, g2, g3, g4, 0);
  };
  if (wave == 0) issue_tile(dir ? (T_SZ - 1) : 0, ldsG0);
#endif
  __syncthreads();

  for (int t = 0; t < T_SZ; ++t) {
    const int ti = dir ? (T_SZ - 1 - t) : t;
    float* ldsG = ldsG0 + (t & 1) * 16 * G;
#if USE_TDM
    if (wave == 0) {
      if (t + 1 < T_SZ) {  // prefetch next tile into the other buffer via TDM
        const int tn = dir ? (T_SZ - 2 - t) : (t + 1);
        issue_tile(tn, ldsG0 + ((t + 1) & 1) * 16 * G);
        __builtin_amdgcn_s_wait_tensorcnt(1);  // current tile landed
      } else {
        __builtin_amdgcn_s_wait_tensorcnt(0);
      }
    }
#else
    for (int idx = tid; idx < 16 * G; idx += 256) {
      int m = idx / G, n = idx - m * G;
      ldsG[idx] = xg[((size_t)(b0 + m) * T_SZ + ti) * G + n];
    }
    if (t + 1 < T_SZ) {
      const int tn = dir ? (T_SZ - 2 - t) : (t + 1);
      __builtin_prefetch(
          xg + ((size_t)(b0 + (tid >> 4)) * T_SZ + tn) * G + (tid & 15) * (G / 16),
          0, 0);
    }
#endif
    __syncthreads();
    // gates += h @ Whh^T via WMMA; C operand comes from the staged tile
#pragma unroll
    for (int tw = 0; tw < TPW; ++tw) {
      const int nt = wave * TPW + tw;
      const int nl = lane & 15, mb = (lane >> 4) * 8;
      v8f acc;
#pragma unroll
      for (int j = 0; j < 8; ++j) acc[j] = ldsG[(mb + j) * G + nt * 16 + nl];
#pragma unroll
      for (int kc = 0; kc < KC; ++kc) {
        v16h a = *(const v16h*)&ldsA[kc * 512 + lane * 16];
        v16h b = *(const v16h*)&ldsB[(nt * KC + kc) * 512 + lane * 16];
        acc = wmma16x16x32(a, b, acc);
      }
#pragma unroll
      for (int j = 0; j < 8; ++j) ldsG[(mb + j) * G + nt * 16 + nl] = acc[j];
    }
    __syncthreads();
    // LSTM cell elementwise; write h straight into A-fragment layout
#pragma unroll
    for (int q = 0; q < CPT; ++q) {
      const int j = pj0 + q * 16;
      float gi = ldsG[pm * G + j];
      float gf = ldsG[pm * G + H + j];
      float gg = ldsG[pm * G + 2 * H + j];
      float go = ldsG[pm * G + 3 * H + j];
      float c = sigmoidf_(gf) * c_state[q] + sigmoidf_(gi) * tanhf(gg);
      c_state[q] = c;
      float hval = sigmoidf_(go) * tanhf(c);
      ldsA[(j >> 5) * 512 + afrag_idx(pm, j & 31)] = (_Float16)hval;
      out[((size_t)(b0 + pm) * T_SZ + ti) * (2 * H) + dir * H + j] = hval;
    }
    __syncthreads();
  }
}

// ============================================================================
// LayerNorm over last dim; one wave32 per row, shfl_xor reductions.
// ============================================================================
__global__ void ln_kernel(float* __restrict__ h, const float* __restrict__ g,
                          const float* __restrict__ be, int width) {
  const int wave = threadIdx.x >> 5, lane = threadIdx.x & 31;
  const size_t row = (size_t)blockIdx.x * 8 + wave;
  float* p = h + row * width;
  const int per = width >> 5;  // <= 8
  float v[8];
  float s = 0.f;
  for (int i = 0; i < per; ++i) { v[i] = p[lane + i * 32]; s += v[i]; }
  for (int m = 16; m >= 1; m >>= 1) s += __shfl_xor(s, m, 32);
  const float mean = s / (float)width;
  float vs = 0.f;
  for (int i = 0; i < per; ++i) { float d = v[i] - mean; vs += d * d; }
  for (int m = 16; m >= 1; m >>= 1) vs += __shfl_xor(vs, m, 32);
  const float rstd = rsqrtf(vs / (float)width + 1e-5f);
  for (int i = 0; i < per; ++i) {
    int col = lane + i * 32;
    p[col] = (v[i] - mean) * rstd * g[col] + be[col];
  }
}

// ============================================================================
// Attention softmax over T + pooled sum + final FC. One block per batch elem.
// ============================================================================
__global__ void attn_fc_kernel(const float* __restrict__ h3,
                               const float* __restrict__ Wa,
                               const float* __restrict__ ba,
                               const float* __restrict__ Wfc,
                               const float* __restrict__ bfc,
                               float* __restrict__ out) {
  __shared__ float lg[T_SZ];
  __shared__ float red[256];
  __shared__ float pooled[64];
  const int tid = threadIdx.x, b = blockIdx.x;
  const float* hb = h3 + (size_t)b * T_SZ * 64;
  float lmax = -3.4e38f;
  for (int t = tid; t < T_SZ; t += 256) {
    float s = ba[0];
#pragma unroll
    for (int d = 0; d < 64; ++d) s += hb[t * 64 + d] * Wa[d];
    lg[t] = s;
    lmax = fmaxf(lmax, s);
  }
  red[tid] = lmax;
  __syncthreads();
  for (int s = 128; s >= 1; s >>= 1) {
    if (tid < s) red[tid] = fmaxf(red[tid], red[tid + s]);
    __syncthreads();
  }
  lmax = red[0];
  __syncthreads();
  float lsum = 0.f;
  for (int t = tid; t < T_SZ; t += 256) {
    float e = __expf(lg[t] - lmax);
    lg[t] = e;
    lsum += e;
  }
  red[tid] = lsum;
  __syncthreads();
  for (int s = 128; s >= 1; s >>= 1) {
    if (tid < s) red[tid] += red[tid + s];
    __syncthreads();
  }
  const float inv = 1.0f / red[0];
  __syncthreads();
  const int d = tid & 63, tg = tid >> 6;
  float acc = 0.f;
  for (int t = tg; t < T_SZ; t += 4) acc += lg[t] * hb[t * 64 + d];
  red[tid] = acc * inv;
  __syncthreads();
  if (tid < 64) pooled[tid] = red[tid] + red[64 + tid] + red[128 + tid] + red[192 + tid];
  __syncthreads();
  if (tid < 4) {
    float s = bfc[tid];
#pragma unroll
    for (int dd = 0; dd < 64; ++dd) s += pooled[dd] * Wfc[tid * 64 + dd];
    out[b * 4 + tid] = s;
  }
}

// ============================================================================
extern "C" void kernel_launch(void* const* d_in, const int* in_sizes, int n_in,
                              void* d_out, int out_size, void* d_ws,
                              size_t ws_size, hipStream_t stream) {
  (void)in_sizes; (void)n_in; (void)out_size; (void)ws_size;
  const float* x      = (const float*)d_in[0];
  const float* W1f_ih = (const float*)d_in[1];
  const float* W1f_hh = (const float*)d_in[2];
  const float* b1f    = (const float*)d_in[3];
  const float* W1b_ih = (const float*)d_in[4];
  const float* W1b_hh = (const float*)d_in[5];
  const float* b1b    = (const float*)d_in[6];
  const float* g1     = (const float*)d_in[7];
  const float* be1    = (const float*)d_in[8];
  const float* W2f_ih = (const float*)d_in[9];
  const float* W2f_hh = (const float*)d_in[10];
  const float* b2f    = (const float*)d_in[11];
  const float* W2b_ih = (const float*)d_in[12];
  const float* W2b_hh = (const float*)d_in[13];
  const float* b2b    = (const float*)d_in[14];
  const float* g2     = (const float*)d_in[15];
  const float* be2    = (const float*)d_in[16];
  const float* W3f_ih = (const float*)d_in[17];
  const float* W3f_hh = (const float*)d_in[18];
  const float* b3f    = (const float*)d_in[19];
  const float* W3b_ih = (const float*)d_in[20];
  const float* W3b_hh = (const float*)d_in[21];
  const float* b3b    = (const float*)d_in[22];
  const float* g3     = (const float*)d_in[23];
  const float* be3    = (const float*)d_in[24];
  const float* Wa     = (const float*)d_in[25];
  const float* ba     = (const float*)d_in[26];
  const float* Wfc    = (const float*)d_in[27];
  const float* bfc    = (const float*)d_in[28];
  float* out = (float*)d_out;

  char* ws = (char*)d_ws;
  size_t off = 0;
  auto alloc = [&](size_t nfloats) {
    float* p = (float*)(ws + off);
    off += nfloats * sizeof(float);
    return p;
  };
  float* xgF = alloc((size_t)B_SZ * T_SZ * 512);
  float* xgB = alloc((size_t)B_SZ * T_SZ * 512);
  float* h1  = alloc((size_t)B_SZ * T_SZ * 256);
  float* h2  = alloc((size_t)B_SZ * T_SZ * 128);
  float* h3  = alloc((size_t)B_SZ * T_SZ * 64);

  const int ldsG1 = 128 * 512 * 2 + 128 * 128 * 4;  // gemm_in<103,512>
  const int ldsG2 = 256 * 256 * 2 + 128 * 256 * 4;  // gemm_in<256,256>
  const int ldsG3 = 128 * 128 * 2 + 128 * 128 * 4;  // gemm_in<128,128>
  const int ldsS1 = 128 * 512 * 2 + 4 * 1024 + 2 * 16 * 512 * 4;  // scan<128>
  const int ldsS2 = 64 * 256 * 2 + 2 * 1024 + 2 * 16 * 256 * 4;   // scan<64>
  const int ldsS3 = 32 * 128 * 2 + 1 * 1024 + 2 * 16 * 128 * 4;   // scan<32>

  (void)hipFuncSetAttribute(reinterpret_cast<const void*>(gemm_in_kernel<103, 512>),
                            hipFuncAttributeMaxDynamicSharedMemorySize, ldsG1);
  (void)hipFuncSetAttribute(reinterpret_cast<const void*>(gemm_in_kernel<256, 256>),
                            hipFuncAttributeMaxDynamicSharedMemorySize, ldsG2);
  (void)hipFuncSetAttribute(reinterpret_cast<const void*>(gemm_in_kernel<128, 128>),
                            hipFuncAttributeMaxDynamicSharedMemorySize, ldsG3);
  (void)hipFuncSetAttribute(reinterpret_cast<const void*>(scan_kernel<128>),
                            hipFuncAttributeMaxDynamicSharedMemorySize, ldsS1);
  (void)hipFuncSetAttribute(reinterpret_cast<const void*>(scan_kernel<64>),
                            hipFuncAttributeMaxDynamicSharedMemorySize, ldsS2);
  (void)hipFuncSetAttribute(reinterpret_cast<const void*>(scan_kernel<32>),
                            hipFuncAttributeMaxDynamicSharedMemorySize, ldsS3);

  const int nRowBlocks = (B_SZ * T_SZ) / 128;  // 1024

  // ---- Layer 1 (din=103, H=128) ----
  gemm_in_kernel<103, 512><<<nRowBlocks, 256, ldsG1, stream>>>(x, W1f_ih, b1f, xgF);
  gemm_in_kernel<103, 512><<<nRowBlocks, 256, ldsG1, stream>>>(x, W1b_ih, b1b, xgB);
  scan_kernel<128><<<dim3(B_SZ / 16, 2), 256, ldsS1, stream>>>(xgF, xgB, W1f_hh, W1b_hh, h1);
  ln_kernel<<<(B_SZ * T_SZ) / 8, 256, 0, stream>>>(h1, g1, be1, 256);

  // ---- Layer 2 (din=256, H=64) ----
  gemm_in_kernel<256, 256><<<nRowBlocks, 256, ldsG2, stream>>>(h1, W2f_ih, b2f, xgF);
  gemm_in_kernel<256, 256><<<nRowBlocks, 256, ldsG2, stream>>>(h1, W2b_ih, b2b, xgB);
  scan_kernel<64><<<dim3(B_SZ / 16, 2), 256, ldsS2, stream>>>(xgF, xgB, W2f_hh, W2b_hh, h2);
  ln_kernel<<<(B_SZ * T_SZ) / 8, 256, 0, stream>>>(h2, g2, be2, 128);

  // ---- Layer 3 (din=128, H=32) ----
  gemm_in_kernel<128, 128><<<nRowBlocks, 256, ldsG3, stream>>>(h2, W3f_ih, b3f, xgF);
  gemm_in_kernel<128, 128><<<nRowBlocks, 256, ldsG3, stream>>>(h2, W3b_ih, b3b, xgB);
  scan_kernel<32><<<dim3(B_SZ / 16, 2), 256, ldsS3, stream>>>(xgF, xgB, W3f_hh, W3b_hh, h3);
  ln_kernel<<<(B_SZ * T_SZ) / 8, 256, 0, stream>>>(h3, g3, be3, 64);

  // ---- Attention + FC ----
  attn_fc_kernel<<<B_SZ, 256, 0, stream>>>(h3, Wa, ba, Wfc, bfc, out);
}
